// MixtralAttention_4861902979224
// MI455X (gfx1250) — 16-bit matrix operand order repaired and checked by reading, not compiled, not run on hardware
//
#include <hip/hip_runtime.h>
#include <hip/hip_bf16.h>

typedef int      v8i  __attribute__((ext_vector_type(8)));
typedef float    v8f  __attribute__((ext_vector_type(8)));
typedef _Float16 v16h __attribute__((ext_vector_type(16)));

union U16H { v16h v; int i[8]; };

#define T_TOK 4096
#define HID   4096
#define NQKV  6144
#define SEQ   2048
#define NKVH  8
#define NH    32
#define HD    128

__device__ __host__ inline int chunk_perm(int w) {
    return (((w & 7) >> 1) * 4) + (w & 1) + ((w >> 3) * 2);
}

__global__ void pack_i32_to_i8(const int* __restrict__ in, int* __restrict__ out, int n4) {
    int i = blockIdx.x * blockDim.x + threadIdx.x;
    if (i >= n4) return;
    const int src = (i & ~15) + chunk_perm(i & 15);
    const int4 a = ((const int4*)in)[src];
    out[i] = (a.x & 255) | ((a.y & 255) << 8) | ((a.z & 255) << 16) | ((a.w & 255) << 24);
}

__device__ inline v8i ld32B(const signed char* p) {
    const int* q = (const int*)p;
    v8i r;
#pragma unroll
    for (int v = 0; v < 8; ++v) r[v] = q[v];
    return r;
}

__device__ inline void i4gemm_tile32(const signed char* __restrict__ A8,
                                     const signed char* __restrict__ W8,
                                     const float* __restrict__ scales,
                                     int t0, int n0, float faccL[8], float faccR[8]) {
    const int lane = threadIdx.x & 31;
    const int nl = lane & 15, h = lane >> 4;
    const signed char* arow  = A8 + (size_t)(t0 + nl) * HID + h * 32;
    const signed char* browL = W8 + (size_t)(n0 + nl) * HID + h * 32;
    const signed char* browR = browL + (size_t)16 * HID;
    const float* scL = scales + (size_t)(n0 + nl) * 32;
    const float* scR = scL + 16 * 32;
#pragma unroll
    for (int r = 0; r < 8; ++r) { faccL[r] = 0.0f; faccR[r] = 0.0f; }

    const v8i zi = {0, 0, 0, 0, 0, 0, 0, 0};
#pragma unroll 2
    for (int g = 0; g < 32; ++g) {
        const int kb = g * 128;
        v8i a0  = ld32B(arow  + kb);  v8i a1  = ld32B(arow  + kb + 64);
        v8i bl0 = ld32B(browL + kb);  v8i bl1 = ld32B(browL + kb + 64);
        v8i br0 = ld32B(browR + kb);  v8i br1 = ld32B(browR + kb + 64);
        v8i i0 = __builtin_amdgcn_wmma_i32_16x16x64_iu8(true, a0, true, bl0, zi, false, false);
        v8i j0 = __builtin_amdgcn_wmma_i32_16x16x64_iu8(true, a0, true, br0, zi, false, false);
        v8i i1 = __builtin_amdgcn_wmma_i32_16x16x64_iu8(true, a1, true, bl1, zi, false, false);
        v8i j1 = __builtin_amdgcn_wmma_i32_16x16x64_iu8(true, a1, true, br1, zi, false, false);
        const float sL = scL[g], sR = scR[g];
#pragma unroll
        for (int r = 0; r < 8; ++r) faccL[r] += ((float)i0[r] + (float)i1[r]) * sL;
#pragma unroll
        for (int r = 0; r < 8; ++r) faccR[r] += ((float)j0[r] + (float)j1[r]) * sR;
    }
}

__global__ void qkv_gemm_kernel(const signed char* __restrict__ A8,
                                const signed char* __restrict__ W8,
                                const float* __restrict__ sqkv,
                                const float* __restrict__ qscale,
                                _Float16* __restrict__ qf, _Float16* __restrict__ kf,
                                _Float16* __restrict__ vf) {
    const int wid = threadIdx.x >> 5;
    const int n0 = (blockIdx.x * 8 + wid) * 32;
    const int t0 = blockIdx.y * 16;
    float faccL[8], faccR[8];
    i4gemm_tile32(A8, W8, sqkv, t0, n0, faccL, faccR);
    const int lane = threadIdx.x & 31;
    const int nl = lane & 15, h = lane >> 4;
#pragma unroll
    for (int sub = 0; sub < 2; ++sub) {
        const int n = n0 + sub * 16 + nl;
        const float* facc = sub ? faccR : faccL;
#pragma unroll
        for (int r = 0; r < 8; ++r) {
            const int row = t0 + r + h * 8;
            const float val = facc[r] * qscale[row];
            if (n < HID)             qf[(size_t)row * HID  + n]          = (_Float16)val;
            else if (n < HID + 1024) kf[(size_t)row * 1024 + (n - HID)]  = (_Float16)val;
            else                     vf[(size_t)row * 1024 + (n - 5120)] = (_Float16)val;
        }
    }
}

__global__ void rope_q_kernel(_Float16* __restrict__ qf) {
    int idx = blockIdx.x * blockDim.x + threadIdx.x;
    if (idx >= T_TOK * NH * 64) return;
    int j = idx & 63; int tmp = idx >> 6; int hh = tmp & 31; int t = tmp >> 5;
    float pos = (float)(t & (SEQ - 1));
    float freq = __powf(1.0e6f, -(float)j * (1.0f / 64.0f));
    float s, c; __sincosf(pos * freq, &s, &c);
    size_t base = (size_t)t * HID + hh * HD + j;
    float x1 = (float)qf[base], x2 = (float)qf[base + 64];
    qf[base]      = (_Float16)(x1 * c - x2 * s);
    qf[base + 64] = (_Float16)(x2 * c + x1 * s);
}

__global__ void rope_k_kernel(_Float16* __restrict__ kf) {
    int idx = blockIdx.x * blockDim.x + threadIdx.x;
    if (idx >= T_TOK * NKVH * 64) return;
    int j = idx & 63; int tmp = idx >> 6; int g = tmp & 7; int t = tmp >> 3;
    float pos = (float)(t & (SEQ - 1));
    float freq = __powf(1.0e6f, -(float)j * (1.0f / 64.0f));
    float s, c; __sincosf(pos * freq, &s, &c);
    size_t base = (size_t)t * 1024 + g * HD + j;
    float x1 = (float)kf[base], x2 = (float)kf[base + 64];
    kf[base]      = (_Float16)(x1 * c - x2 * s);
    kf[base + 64] = (_Float16)(x2 * c + x1 * s);
}

__global__ void vtrans_kernel(const _Float16* __restrict__ vf, _Float16* __restrict__ vt) {
    int idx = blockIdx.x * blockDim.x + threadIdx.x;
    if (idx >= T_TOK * 1024) return;
    int d = idx & 127; int tmp = idx >> 7; int g = tmp & 7; int t = tmp >> 3;
    int b = t >> 11, s = t & (SEQ - 1);
    vt[((size_t)(b * NKVH + g) * HD + d) * SEQ + s] = vf[(size_t)t * 1024 + g * HD + d];
}

__global__ __launch_bounds__(256) void flash_attn_kernel(const _Float16* __restrict__ qf,
                                                         const _Float16* __restrict__ kf,
                                                         const _Float16* __restrict__ vt,
                                                         float* __restrict__ attn) {
    __shared__ _Float16 pLds[8 * 16 * 32];
    const int lane = threadIdx.x & 31;
    const int wid  = threadIdx.x >> 5;
    int w = blockIdx.x * 8 + wid;
    const int qblk = w & 127; w >>= 7;
    const int hq   = w & 3;   w >>= 2;
    const int g    = w & 7;   const int b = w >> 3;
    const int head = g * 4 + hq;
    const int nl = lane & 15, hh = lane >> 4;
    const int t0 = b * SEQ + qblk * 16;

    U16H aQ[4];
    const _Float16* qrow = qf + (size_t)(t0 + nl) * HID + head * HD;
#pragma unroll
    for (int c = 0; c < 4; ++c) {
        const int* qp = (const int*)(qrow + c * 32 + hh * 16);
#pragma unroll
        for (int v = 0; v < 8; ++v) aQ[c].i[v] = qp[v];
    }

    const v8f zero8 = {0, 0, 0, 0, 0, 0, 0, 0};
    v8f O[8];
#pragma unroll
    for (int dc = 0; dc < 8; ++dc) O[dc] = zero8;
    float mrow[8], lrow[8];
#pragma unroll
    for (int r = 0; r < 8; ++r) { mrow[r] = -3.0e38f; lrow[r] = 0.0f; }

    _Float16* pl = pLds + wid * 512;
    const float scale = 0.08838834764831845f;
    const _Float16* vtb = vt + (size_t)(b * NKVH + g) * HD * SEQ;
    const int nsteps = qblk / 2 + 1;

    for (int st = 0; st < nsteps; ++st) {
        const int k0 = st * 32;
        if (st + 1 < nsteps) {
            __builtin_prefetch(kf + (size_t)(b * SEQ + k0 + 32 + lane) * 1024 + g * HD, 0, 3);
            __builtin_prefetch(vtb + (size_t)(lane * 4) * SEQ + k0 + 32, 0, 3);
        }
        v8f s0 = zero8, s1 = zero8;
#pragma unroll
        for (int c = 0; c < 4; ++c) {
            U16H b0, b1;
            const int* p0 = (const int*)(kf + (size_t)(b * SEQ + k0 + nl) * 1024 + g * HD + c * 32 + hh * 16);
            const int* p1 = p0 + (16 * 1024) / 2;
#pragma unroll
            for (int v = 0; v < 8; ++v) { b0.i[v] = p0[v]; b1.i[v] = p1[v]; }
            s0 = __builtin_amdgcn_wmma_f32_16x16x32_f16(false, aQ[c].v, false, b0.v, (short)0, s0, false, false);
            s1 = __builtin_amdgcn_wmma_f32_16x16x32_f16(false, aQ[c].v, false, b1.v, (short)0, s1, false, false);
        }
        float p0a[8], p1a[8], fac[8];
#pragma unroll
        for (int r = 0; r < 8; ++r) {
            const int qi  = qblk * 16 + r + hh * 8;
            const int k0i = k0 + nl, k1i = k0 + 16 + nl;
            float v0 = (k0i <= qi) ? s0[r] * scale : -3.0e38f;
            float v1 = (k1i <= qi) ? s1[r] * scale : -3.0e38f;
            float m = fmaxf(v0, v1);
            m = fmaxf(m, __shfl_xor(m, 1)); m = fmaxf(m, __shfl_xor(m, 2));
            m = fmaxf(m, __shfl_xor(m, 4)); m = fmaxf(m, __shfl_xor(m, 8));
            const float mn = fmaxf(mrow[r], m);
            fac[r] = __expf(mrow[r] - mn);
            v0 = __expf(v0 - mn); v1 = __expf(v1 - mn);
            p0a[r] = v0; p1a[r] = v1;
            float rs = v0 + v1;
            rs += __shfl_xor(rs, 1); rs += __shfl_xor(rs, 2);
            rs += __shfl_xor(rs, 4); rs += __shfl_xor(rs, 8);
            lrow[r] = lrow[r] * fac[r] + rs;
            mrow[r] = mn;
        }
#pragma unroll
        for (int dc = 0; dc < 8; ++dc)
#pragma unroll
            for (int r = 0; r < 8; ++r) O[dc][r] *= fac[r];
#pragma unroll
        for (int r = 0; r < 8; ++r) {
            pl[(r + hh * 8) * 32 + nl]      = (_Float16)p0a[r];
            pl[(r + hh * 8) * 32 + 16 + nl] = (_Float16)p1a[r];
        }
        asm volatile("" ::: "memory");
        U16H aP;
#pragma unroll
        for (int v = 0; v < 8; ++v) {
            const int K = hh * 16 + v * 2;
            aP.i[v] = *(const int*)(pl + nl * 32 + K);
        }
        asm volatile("" ::: "memory");
#pragma unroll
        for (int dc = 0; dc < 8; ++dc) {
            U16H bv;
            const int* vp = (const int*)(vtb + (size_t)(dc * 16 + nl) * SEQ + k0 + hh * 16);
#pragma unroll
            for (int v = 0; v < 8; ++v) bv.i[v] = vp[v];
            O[dc] = __builtin_amdgcn_wmma_f32_16x16x32_f16(false, aP.v, false, bv.v, (short)0, O[dc], false, false);
        }
    }
#pragma unroll
    for (int r = 0; r < 8; ++r) {
        const float rl = 1.0f / lrow[r];
        const int row = t0 + r + hh * 8;
#pragma unroll
        for (int dc = 0; dc < 8; ++dc)
            attn[(size_t)row * HID + head * HD + dc * 16 + nl] = O[dc][r] * rl;
    }
}

__global__ void quant_kernel(const float* __restrict__ attn, int* __restrict__ aq,
                             float* __restrict__ ascale) {
    __shared__ float red[256];
    const int t = blockIdx.x;
    const float* row = attn + (size_t)t * HID;
    float amax = 0.0f;
    for (int i = threadIdx.x; i < HID; i += 256) amax = fmaxf(amax, fabsf(row[i]));
    red[threadIdx.x] = amax;
    __syncthreads();
    for (int s = 128; s > 0; s >>= 1) {
        if (threadIdx.x < s) red[threadIdx.x] = fmaxf(red[threadIdx.x], red[threadIdx.x + s]);
        __syncthreads();
    }
    const float am = fmaxf(red[0], 1e-6f);
    const float sc = am / 127.0f, inv = 127.0f / am;
    if (threadIdx.x == 0) ascale[t] = sc;
    for (int i = threadIdx.x; i < HID / 4; i += 256) {
        const int src4 = ((i & ~15) + chunk_perm(i & 15)) * 4;
        int pk = 0;
#pragma unroll
        for (int bpos = 0; bpos < 4; ++bpos) {
            float qv = rintf(row[src4 + bpos] * inv);
            qv = fminf(fmaxf(qv, -127.0f), 127.0f);
            pk |= ((int)qv & 255) << (bpos * 8);
        }
        aq[(size_t)t * (HID / 4) + i] = pk;
    }
}

__global__ void out_gemm_kernel(const signed char* __restrict__ A8,
                                const signed char* __restrict__ W8,
                                const float* __restrict__ so,
                                const float* __restrict__ ascale,
                                float* __restrict__ out) {
    const int wid = threadIdx.x >> 5;
    const int n0 = (blockIdx.x * 8 + wid) * 32;
    const int t0 = blockIdx.y * 16;
    float faccL[8], faccR[8];
    i4gemm_tile32(A8, W8, so, t0, n0, faccL, faccR);
    const int lane = threadIdx.x & 31;
    const int nl = lane & 15, h = lane >> 4;
#pragma unroll
    for (int sub = 0; sub < 2; ++sub) {
        const int n = n0 + sub * 16 + nl;
        const float* facc = sub ? faccR : faccL;
#pragma unroll
        for (int r = 0; r < 8; ++r) {
            const int row = t0 + r + h * 8;
            out[(size_t)row * HID + n] = facc[r] * ascale[row];
        }
    }
}

extern "C" void kernel_launch(void* const* d_in, const int* in_sizes, int n_in,
                              void* d_out, int out_size, void* d_ws, size_t ws_size,
                              hipStream_t stream) {
    const int*   q_hidden = (const int*)d_in[0];
    const float* q_scale  = (const float*)d_in[1];
    const int*   w_qkv    = (const int*)d_in[3];
    const float* s_qkv    = (const float*)d_in[4];
    const int*   w_o      = (const int*)d_in[5];
    const float* s_o      = (const float*)d_in[6];
    float* out = (float*)d_out;

    char* ws = (char*)d_ws;
    size_t o = 0;
    auto alloc = [&](size_t bytes) { o = (o + 255) & ~(size_t)255; size_t r = o; o += bytes; return r; };
    signed char* q8    = (signed char*)(ws + alloc((size_t)T_TOK * HID));
    signed char* wq8   = (signed char*)(ws + alloc((size_t)NQKV * HID));
    signed char* wo8   = (signed char*)(ws + alloc((size_t)HID * HID));
    _Float16*    qf    = (_Float16*)(ws + alloc((size_t)T_TOK * HID * 2));
    _Float16*    kf    = (_Float16*)(ws + alloc((size_t)T_TOK * 1024 * 2));
    _Float16*    vf    = (_Float16*)(ws + alloc((size_t)T_TOK * 1024 * 2));
    _Float16*    vt    = (_Float16*)(ws + alloc((size_t)T_TOK * 1024 * 2));
    float*       attn  = (float*)(ws + alloc((size_t)T_TOK * HID * 4));
    signed char* aq8   = (signed char*)(ws + alloc((size_t)T_TOK * HID));
    float*       asc   = (float*)(ws + alloc((size_t)T_TOK * 4));
    (void)ws_size; (void)n_in; (void)in_sizes; (void)out_size;

    pack_i32_to_i8<<<(T_TOK * HID / 4) / 256, 256, 0, stream>>>(q_hidden, (int*)q8, T_TOK * HID / 4);
    pack_i32_to_i8<<<(NQKV * HID / 4) / 256, 256, 0, stream>>>(w_qkv, (int*)wq8, NQKV * HID / 4);
    pack_i32_to_i8<<<(HID * HID / 4) / 256, 256, 0, stream>>>(w_o, (int*)wo8, HID * HID / 4);

    qkv_gemm_kernel<<<dim3(NQKV / 256, T_TOK / 16), 256, 0, stream>>>(q8, wq8, s_qkv, q_scale, qf, kf, vf);

    rope_q_kernel<<<(T_TOK * NH * 64) / 256, 256, 0, stream>>>(qf);
    rope_k_kernel<<<(T_TOK * NKVH * 64) / 256, 256, 0, stream>>>(kf);
    vtrans_kernel<<<(T_TOK * 1024) / 256, 256, 0, stream>>>(vf, vt);

    flash_attn_kernel<<<1024, 256, 0, stream>>>(qf, kf, vt, attn);

    quant_kernel<<<T_TOK, 256, 0, stream>>>(attn, (int*)aq8, asc);

    out_gemm_kernel<<<dim3(HID / 256, T_TOK / 16), 256, 0, stream>>>(aq8, wo8, s_o, asc, out);
}
